// GNNRecommendationModel_89524298318419
// MI455X (gfx1250) — matrix operand order repaired and checked by reading, not compiled, not run on hardware
//
#include <hip/hip_runtime.h>
#include <hip/hip_bf16.h>

#define N_NODES 100000
#define N_EDGES 1000000
#define HID     64

typedef _Float16 v16h __attribute__((ext_vector_type(16)));
typedef _Float16 v8h  __attribute__((ext_vector_type(8)));
typedef float    v8f  __attribute__((ext_vector_type(8)));

// ---------------- degree / norm ----------------
__global__ void k_init_deg(float* deg) {
    int n = blockIdx.x * blockDim.x + threadIdx.x;
    if (n < N_NODES) deg[n] = 1.0f;            // self-loop weight
}

__global__ void k_deg_edges(const long long* ei, const float* ew, float* deg) {
    int e = blockIdx.x * blockDim.x + threadIdx.x;
    if (e < N_EDGES) {
        int c = (int)ei[N_EDGES + e];          // col
        atomicAdd(&deg[c], ew[e]);
    }
}

__global__ void k_dinv(float* deg) {
    int n = blockIdx.x * blockDim.x + threadIdx.x;
    if (n < N_NODES) {
        float d = deg[n];
        deg[n] = (d > 0.0f) ? rsqrtf(d) : 0.0f;   // in-place -> dinv
    }
}

// ------- x@W (skinny) + self-loop contribution into fused h buffer -------
__global__ void k_xw_self(const float* __restrict__ ux, const float* __restrict__ mx,
                          const float* __restrict__ Wu, const float* __restrict__ Wm,
                          const float* __restrict__ dinv,
                          float* __restrict__ xwu, float* __restrict__ xwm,
                          float* __restrict__ hbuf) {
    int t = blockIdx.x * blockDim.x + threadIdx.x;
    if (t >= N_NODES * HID) return;
    int n = t >> 6, h = t & 63;
    float su = 0.0f;
#pragma unroll
    for (int k = 0; k < 3; ++k)  su += ux[n * 3 + k] * Wu[k * HID + h];
    float sm = 0.0f;
#pragma unroll
    for (int k = 0; k < 18; ++k) sm += mx[n * 18 + k] * Wm[k * HID + h];
    xwu[t] = su;
    xwm[t] = sm;
    float di = dinv[n];
    float sn = di * di;                         // self-loop norm: dinv*1*dinv
    hbuf[n * 128 + h]      = sn * su;           // user half of concat
    hbuf[n * 128 + 64 + h] = sn * sm;           // movie half of concat
}

// ------- edge scatter: 64 lanes per edge, both convs fused -------
__global__ void k_scatter(const long long* __restrict__ ei, const float* __restrict__ ew,
                          const float* __restrict__ dinv,
                          const float* __restrict__ xwu, const float* __restrict__ xwm,
                          float* __restrict__ hbuf) {
    long long t = (long long)blockIdx.x * blockDim.x + threadIdx.x;
    if (t >= (long long)N_EDGES * HID) return;
    int e = (int)(t >> 6);
    int h = (int)(t & 63);
    int r = (int)ei[e];
    int c = (int)ei[N_EDGES + e];
    float nrm = dinv[r] * ew[e] * dinv[c];
    atomicAdd(&hbuf[(long long)c * 128 + h],      nrm * xwu[(long long)r * 64 + h]);
    atomicAdd(&hbuf[(long long)c * 128 + 64 + h], nrm * xwm[(long long)r * 64 + h]);
}

// ------- pack W1 (f32 [128,128]) into per-lane WMMA B-fragment order, f16 -------
// w1p[((t*8 + c)*32 + lane)*16 + i] = W1[(t*32 + (lane<16?0:8) + (i&7) + (i<8?0:16))*128 + c*16 + (lane&15)]
__global__ void k_pack_w1(const float* __restrict__ W1, _Float16* __restrict__ w1p) {
    int i = blockIdx.x * blockDim.x + threadIdx.x;
    if (i >= 128 * 128) return;
    int ii   = i & 15;
    int lane = (i >> 4) & 31;
    int c    = (i >> 9) & 7;
    int t    = (i >> 12);
    int k    = t * 32 + ((lane < 16) ? 0 : 8) + (ii & 7) + ((ii < 8) ? 0 : 16);
    int col  = c * 16 + (lane & 15);
    w1p[i] = (_Float16)W1[k * 128 + col];
}

// ------- fold GCN bias into h, convert to f16 row-major [N,128] -------
__global__ void k_bias_h(const float* __restrict__ hbuf,
                         const float* __restrict__ b_user, const float* __restrict__ b_movie,
                         _Float16* __restrict__ hf) {
    long long i = (long long)blockIdx.x * blockDim.x + threadIdx.x;
    if (i >= (long long)N_NODES * 128) return;
    int k = (int)(i & 127);
    float bias = (k < 64) ? b_user[k] : b_movie[k - 64];
    hf[i] = (_Float16)(hbuf[i] + bias);
}

// ------- MLP: hf16[N,128] -> relu(h@W1+b1) -> @W2+b2, WMMA f16/f32 -------
__global__ __launch_bounds__(256)
void k_mlp(const _Float16* __restrict__ hf, const _Float16* __restrict__ w1p,
           const float* __restrict__ b1, const float* __restrict__ W2,
           const float* __restrict__ b2, float* __restrict__ out) {
    int wave = blockIdx.x * (blockDim.x >> 5) + (threadIdx.x >> 5);
    if (wave >= N_NODES / 16) return;          // wave-uniform; EXEC stays full
    int lane = threadIdx.x & 31;
    int m    = lane & 15;                      // A row / B,C col within tile
    int half = lane >> 4;                      // 0: lanes 0-15, 1: lanes 16-31
    int kbA  = half * 8;                       // A: hi half holds K+8
    int row  = wave * 16 + m;

    const _Float16* hrow = hf + (long long)row * 128;

    v8f acc[8] = {};                           // 8 column tiles of 16

#pragma unroll
    for (int t = 0; t < 4; ++t) {              // K steps of 32 (K=128)
        // A fragment: two contiguous 8-half chunks at k = t*32+kbA and +16
        v8h lo = *(const v8h*)(hrow + t * 32 + kbA);
        v8h hi = *(const v8h*)(hrow + t * 32 + kbA + 16);
        v16h a;
#pragma unroll
        for (int i = 0; i < 8; ++i) { a[i] = lo[i]; a[i + 8] = hi[i]; }
#pragma unroll
        for (int c = 0; c < 8; ++c) {
            // B fragment: one 32-byte contiguous vector per lane (pre-packed)
            v16h b = *(const v16h*)(w1p + (((t * 8 + c) * 32 + lane) << 4));
            acc[c] = __builtin_amdgcn_wmma_f32_16x16x32_f16(
                false, a, false, b, (short)0, acc[c], false, false);
        }
    }

    // second layer: pred = relu(h1) @ W2 + b2  (per-lane dot + 16-lane reduce)
    float b1v[8], w2v[8];
#pragma unroll
    for (int c = 0; c < 8; ++c) {
        b1v[c] = b1[c * 16 + m];
        w2v[c] = W2[c * 16 + m];
    }
    float b2v = b2[0];
#pragma unroll
    for (int r = 0; r < 8; ++r) {              // C VGPR r -> row r (+8 for hi half)
        float v = 0.0f;
#pragma unroll
        for (int c = 0; c < 8; ++c) {
            float x = acc[c][r] + b1v[c];
            x = x > 0.0f ? x : 0.0f;
            v += x * w2v[c];
        }
#pragma unroll
        for (int off = 1; off < 16; off <<= 1)
            v += __shfl_xor(v, off, 32);       // reduce within each 16-lane group
        if (m == 0)
            out[wave * 16 + half * 8 + r] = v + b2v;
    }
}

extern "C" void kernel_launch(void* const* d_in, const int* in_sizes, int n_in,
                              void* d_out, int out_size, void* d_ws, size_t ws_size,
                              hipStream_t stream) {
    const float*     user_x  = (const float*)d_in[0];
    const float*     movie_x = (const float*)d_in[1];
    const long long* ei      = (const long long*)d_in[2];   // int64 [2,E]
    const float*     ew      = (const float*)d_in[3];
    const float*     W_user  = (const float*)d_in[4];
    const float*     b_user  = (const float*)d_in[5];
    const float*     W_movie = (const float*)d_in[6];
    const float*     b_movie = (const float*)d_in[7];
    const float*     W1      = (const float*)d_in[8];
    const float*     b1      = (const float*)d_in[9];
    const float*     W2      = (const float*)d_in[10];
    const float*     b2      = (const float*)d_in[11];
    float*           out     = (float*)d_out;

    // workspace carve (all buffers written before read each call)
    char*  ws  = (char*)d_ws;
    size_t off = 0;
    auto carve = [&](size_t bytes) {
        void* p = ws + off;
        off = (off + bytes + 255) & ~(size_t)255;
        return p;
    };
    float*    deg  = (float*)carve((size_t)N_NODES * 4);            // then dinv
    float*    xwu  = (float*)carve((size_t)N_NODES * HID * 4);
    float*    xwm  = (float*)carve((size_t)N_NODES * HID * 4);
    float*    hbuf = (float*)carve((size_t)N_NODES * 128 * 4);
    _Float16* hf   = (_Float16*)carve((size_t)N_NODES * 128 * 2);
    _Float16* w1p  = (_Float16*)carve((size_t)128 * 128 * 2);

    k_init_deg <<<(N_NODES + 255) / 256, 256, 0, stream>>>(deg);
    k_deg_edges<<<(N_EDGES + 255) / 256, 256, 0, stream>>>(ei, ew, deg);
    k_dinv     <<<(N_NODES + 255) / 256, 256, 0, stream>>>(deg);
    k_xw_self  <<<(N_NODES * HID + 255) / 256, 256, 0, stream>>>(
        user_x, movie_x, W_user, W_movie, deg, xwu, xwm, hbuf);
    long long scat = (long long)N_EDGES * HID;
    k_scatter  <<<(unsigned)((scat + 255) / 256), 256, 0, stream>>>(
        ei, ew, deg, xwu, xwm, hbuf);
    k_pack_w1  <<<(128 * 128 + 255) / 256, 256, 0, stream>>>(W1, w1p);
    k_bias_h   <<<(int)(((long long)N_NODES * 128 + 255) / 256), 256, 0, stream>>>(
        hbuf, b_user, b_movie, hf);
    k_mlp      <<<(N_NODES / 16 + 7) / 8, 256, 0, stream>>>(
        hf, w1p, b1, W2, b2, out);
}
